// FractalAttentionLayer_54228257079871
// MI455X (gfx1250) — matrix operand order repaired and checked by reading, not compiled, not run on hardware
//
#include <hip/hip_runtime.h>
#include <hip/hip_bf16.h>
#include <math.h>

// ---------------- problem constants ----------------
#define B_N    2
#define S_LEN  2048
#define E_DIM  1024
#define H_N    16
#define D_HEAD 64
#define M_ROWS (B_N * S_LEN)          // 4096
#define KAPPA  0.3f

typedef __bf16 bf16_t;
typedef __attribute__((ext_vector_type(16))) __bf16 bf16x16;
typedef __attribute__((ext_vector_type(8)))  __bf16 bf16x8;
typedef __attribute__((ext_vector_type(8)))  float  f32x8;

// Assemble a 16x32 bf16 A-fragment (contiguous K order, matching the B fragments):
// lane L: m = L%16, h = L/16; elems 0..15 -> K = k0+16h+e
static __device__ inline bf16x16 make_a_frag(const bf16_t* rowptr_k0, int lh) {
    bf16x8 lo = *(const bf16x8*)(rowptr_k0 + 16 * lh);
    bf16x8 hi = *(const bf16x8*)(rowptr_k0 + 16 * lh + 8);
    bf16x16 a;
#pragma unroll
    for (int e = 0; e < 8; ++e) { a[e] = lo[e]; a[e + 8] = hi[e]; }
    return a;
}

static __device__ inline f32x8 wmma_bf16(bf16x16 a, bf16x16 b, f32x8 c) {
    return __builtin_amdgcn_wmma_f32_16x16x32_bf16(false, a, false, b, (short)0, c, false, false);
}

// ---------------- conversion / prep kernels ----------------
__global__ void cvt_bf16_kernel(const float* __restrict__ src, bf16_t* __restrict__ dst, int n) {
    int i = blockIdx.x * blockDim.x + threadIdx.x;
    if (i < n) dst[i] = (bf16_t)src[i];
}

// Wt[n*E + k] = W[k*E + n]   (transpose so the GEMM B-fragment reads contiguous K)
__global__ void cvt_wT_kernel(const float* __restrict__ W, bf16_t* __restrict__ Wt) {
    int i = blockIdx.x * blockDim.x + threadIdx.x;   // i = n*E + k
    int k = i & (E_DIM - 1);
    int n = i >> 10;
    Wt[i] = (bf16_t)W[k * E_DIM + n];
}

// T[(b*S+s)*H + h] = cos((phase[b,s] - carrier[h]) / 2)^2
__global__ void phase_T_kernel(const float* __restrict__ phase, const float* __restrict__ carrier,
                               float* __restrict__ T) {
    int i = blockIdx.x * blockDim.x + threadIdx.x;   // i = (b*S+s)*H + h
    if (i >= B_N * S_LEN * H_N) return;
    int h  = i & (H_N - 1);
    int bs = i >> 4;
    float c = cosf((phase[bs] - carrier[h]) * 0.5f);
    T[i] = c * c;
}

// ---------------- WMMA GEMM (M=4096, N=K=1024), wave tile = 16M x 64N ----------------
// Software-pipelined: k-step t+1 fragments are loaded (into registers that are
// forced live alongside step-t fragments) before the step-t wmma issue, so the
// VMEM latency is hidden under the matrix pipe instead of s_wait_loadcnt 0.
// mode 0: Q    -> bf16 [B,H,S,D], scaled by 1/sqrt(D)
// mode 1: Kc   -> bf16 [B,H,S,D], fractal compressed
// mode 2: Vc   -> bf16 [B,H,D,S] (transposed), fractal compressed
// mode 3: out  -> fp32 [M,E]
__global__ __launch_bounds__(128)
void gemm16_kernel(const bf16_t* __restrict__ A, const bf16_t* __restrict__ Bt,
                   const float* __restrict__ bias, void* __restrict__ dst, int mode) {
    const int lane = threadIdx.x & 31;
    const int wave = threadIdx.x >> 5;
    const int lm = lane & 15, lh = lane >> 4;
    const int mt = blockIdx.x * 16;                       // 16 rows per block
    const int nt = (blockIdx.y * 4 + wave) * 64;          // 64 cols per wave

    const bf16_t* arow = A + (size_t)(mt + lm) * E_DIM;   // lane's A row (m = lm)
    const bf16_t* brow[4];
#pragma unroll
    for (int c = 0; c < 4; ++c)                           // lane's B col rows (n = nt+16c+lm)
        brow[c] = Bt + (size_t)(nt + 16 * c + lm) * E_DIM + 16 * lh;

    f32x8 acc[4] = {};

    // ---- prologue: load k-step 0 ----
    bf16x16 af = make_a_frag(arow, lh);
    bf16x16 bfr[4];
#pragma unroll
    for (int c = 0; c < 4; ++c)
        bfr[c] = *(const bf16x16*)(brow[c]);

    // ---- steady state: load k+32 while computing k ----
#pragma unroll 2
    for (int k0 = 0; k0 < E_DIM - 32; k0 += 32) {
        const int kn = k0 + 32;
        bf16x16 afn = make_a_frag(arow + kn, lh);
        bf16x16 bfn[4];
#pragma unroll
        for (int c = 0; c < 4; ++c)
            bfn[c] = *(const bf16x16*)(brow[c] + kn);      // elems e -> K = kn+16*lh+e
        __builtin_prefetch(arow + kn + 96, 0, 1);          // global_prefetch_b8
        __builtin_prefetch(brow[1] + kn + 96, 0, 1);
        __builtin_prefetch(brow[3] + kn + 96, 0, 1);
#pragma unroll
        for (int c = 0; c < 4; ++c)
            acc[c] = wmma_bf16(af, bfr[c], acc[c]);
        af = afn;
#pragma unroll
        for (int c = 0; c < 4; ++c) bfr[c] = bfn[c];
    }
    // ---- epilogue: last k-step ----
#pragma unroll
    for (int c = 0; c < 4; ++c)
        acc[c] = wmma_bf16(af, bfr[c], acc[c]);

#pragma unroll
    for (int c = 0; c < 4; ++c) {
        const int ng = nt + 16 * c + lm;                  // global col
        const float bv = bias[ng];
#pragma unroll
        for (int r = 0; r < 8; ++r) {
            const int mg = mt + r + 8 * lh;               // global row (token index)
            float v = acc[c][r] + bv;
            if (mode == 3) {
                ((float*)dst)[(size_t)mg * E_DIM + ng] = v;
            } else {
                const int bb = mg >> 11, ss = mg & (S_LEN - 1);
                const int hh = ng >> 6,  dd = ng & (D_HEAD - 1);
                if (mode == 0) {
                    ((bf16_t*)dst)[((size_t)(bb * H_N + hh) * S_LEN + ss) * D_HEAD + dd] =
                        (bf16_t)(v * 0.125f);             // fold 1/sqrt(64)
                } else {
                    const float ex  = 1.0f + KAPPA * (1.0f - (float)ss * (1.0f / (S_LEN - 1)));
                    const float mag = fmaxf(fabsf(v), 1e-8f);
                    const float cv  = copysignf(expf(ex * logf(mag)), v);
                    if (mode == 1)
                        ((bf16_t*)dst)[((size_t)(bb * H_N + hh) * S_LEN + ss) * D_HEAD + dd] = (bf16_t)cv;
                    else
                        ((bf16_t*)dst)[((size_t)(bb * H_N + hh) * D_HEAD + dd) * S_LEN + ss] = (bf16_t)cv;
                }
            }
        }
    }
}

// ---------------- flash attention: 1 wave = 16 queries of one (b,h) ----------------
__global__ __launch_bounds__(128)
void attn_kernel(const bf16_t* __restrict__ Qb, const bf16_t* __restrict__ KCb,
                 const bf16_t* __restrict__ VCt, const float* __restrict__ Tm,
                 const float* __restrict__ lock_raw, bf16_t* __restrict__ AttOut) {
    __shared__ __align__(64) __bf16 ldsP[4][16 * 32];   // per-wave P tile (C->A layout bounce)

    const int lane = threadIdx.x & 31;
    const int wave = threadIdx.x >> 5;
    const int lm = lane & 15, lh = lane >> 4;
    const int it = blockIdx.x * 4 + wave;     // i-tile 0..127
    const int h  = blockIdx.y;
    const int b  = blockIdx.z;
    const int ibase = it * 16;

    const float lock = 1.0f / (1.0f + expf(-lock_raw[h]));
    const float c0   = KAPPA * lock;

    const bf16_t* qhead = Qb  + ((size_t)(b * H_N + h) * S_LEN) * D_HEAD;
    const bf16_t* khead = KCb + ((size_t)(b * H_N + h) * S_LEN) * D_HEAD;
    const bf16_t* vhead = VCt + ((size_t)(b * H_N + h) * D_HEAD) * S_LEN;
    const float*  tbh   = Tm  + (size_t)b * S_LEN * H_N + h;   // stride H per token

    // Q A-fragments for K(=d) 0..31 and 32..63 (Q already scaled by 1/sqrt(D))
    const bf16_t* qrow = qhead + (size_t)(ibase + lm) * D_HEAD;
    bf16x16 aq0 = make_a_frag(qrow,      lh);
    bf16x16 aq1 = make_a_frag(qrow + 32, lh);

    float Ti[8];
#pragma unroll
    for (int r = 0; r < 8; ++r) Ti[r] = tbh[(size_t)(ibase + r + 8 * lh) * H_N];

    f32x8 o[4] = {};                   // 16x64 output accumulator (4 d-tiles)
    float mrun[8], lrun[8];
#pragma unroll
    for (int r = 0; r < 8; ++r) { mrun[r] = -3.0e38f; lrun[r] = 0.0f; }

    const int nblocks = it / 2 + 1;    // 32-wide key blocks up to the diagonal
    for (int t = 0; t < nblocks; ++t) {
        const int jb = t * 32;
        if (t + 1 < nblocks) {         // prefetch next key block (global_prefetch_b8)
            __builtin_prefetch(khead + (size_t)(jb + 32 + lane) * D_HEAD, 0, 1);
            __builtin_prefetch(vhead + (size_t)lane * S_LEN + jb + 32, 0, 1);
        }
        // ---- issue all K-fragment loads for both column groups first ----
        const int j0 = jb + lm;                  // group 0 key column
        const int j1 = jb + 16 + lm;             // group 1 key column
        const bf16_t* krow0 = khead + (size_t)j0 * D_HEAD + 16 * lh;
        const bf16_t* krow1 = khead + (size_t)j1 * D_HEAD + 16 * lh;
        bf16x16 bk00 = *(const bf16x16*)(krow0);        // g0, d = 16*lh + e
        bf16x16 bk01 = *(const bf16x16*)(krow0 + 32);   // g0, d = 32 + ...
        bf16x16 bk10 = *(const bf16x16*)(krow1);        // g1
        bf16x16 bk11 = *(const bf16x16*)(krow1 + 32);
        float tj0 = tbh[(size_t)j0 * H_N];
        float tj1 = tbh[(size_t)j1 * H_N];

        f32x8 s[2] = {};
        s[0] = wmma_bf16(aq0, bk00, s[0]);
        s[1] = wmma_bf16(aq0, bk10, s[1]);
        s[0] = wmma_bf16(aq1, bk01, s[0]);
        s[1] = wmma_bf16(aq1, bk11, s[1]);

        // bonus + causal mask
#pragma unroll
        for (int r = 0; r < 8; ++r) {
            const int ii = ibase + r + 8 * lh;
            float v0 = s[0][r] + c0 * Ti[r] * tj0;
            float v1 = s[1][r] + c0 * Ti[r] * tj1;
            s[0][r] = (j0 <= ii) ? v0 : -3.0e38f;
            s[1][r] = (j1 <= ii) ? v1 : -3.0e38f;
        }

        // online softmax: rows live across the 16 lanes of each half-wave
        float alpha[8];
#pragma unroll
        for (int r = 0; r < 8; ++r) {
            float mx = fmaxf(s[0][r], s[1][r]);
            mx = fmaxf(mx, __shfl_xor(mx, 1, 32));
            mx = fmaxf(mx, __shfl_xor(mx, 2, 32));
            mx = fmaxf(mx, __shfl_xor(mx, 4, 32));
            mx = fmaxf(mx, __shfl_xor(mx, 8, 32));
            const float mnew = fmaxf(mrun[r], mx);
            const float al   = expf(mrun[r] - mnew);
            mrun[r] = mnew;
            float p0 = expf(s[0][r] - mnew);
            float p1 = expf(s[1][r] - mnew);
            s[0][r] = p0; s[1][r] = p1;
            float rs = p0 + p1;
            rs += __shfl_xor(rs, 1, 32);
            rs += __shfl_xor(rs, 2, 32);
            rs += __shfl_xor(rs, 4, 32);
            rs += __shfl_xor(rs, 8, 32);
            lrun[r] = lrun[r] * al + rs;
            alpha[r] = al;
        }
#pragma unroll
        for (int d = 0; d < 4; ++d)
#pragma unroll
            for (int r = 0; r < 8; ++r) o[d][r] *= alpha[r];

        // bounce P through LDS: C-layout (lane=col) -> A-layout (lane=row)
        __bf16* lp = ldsP[wave];
#pragma unroll
        for (int g = 0; g < 2; ++g)
#pragma unroll
            for (int r = 0; r < 8; ++r)
                lp[(r + 8 * lh) * 32 + 16 * g + lm] = (bf16_t)s[g][r];
        asm volatile("s_wait_dscnt 0" ::: "memory");       // CDNA5 split DS counter wait
        bf16x16 ap = make_a_frag(lp + lm * 32, lh);

        // O += P @ Vc   (Vc transposed: row d, contiguous j) — loads first, then wmma
        bf16x16 bv_[4];
#pragma unroll
        for (int d = 0; d < 4; ++d)
            bv_[d] = *(const bf16x16*)(vhead + (size_t)(16 * d + lm) * S_LEN + jb + 16 * lh);
#pragma unroll
        for (int d = 0; d < 4; ++d)
            o[d] = wmma_bf16(ap, bv_[d], o[d]);
    }

    // epilogue: 1/l and gate folded together, store bf16 to [M, E] for the O-GEMM
    float sc[8];
#pragma unroll
    for (int r = 0; r < 8; ++r) {
        const float gate = (1.0f - lock) + lock * Ti[r];
        sc[r] = gate / lrun[r];
    }
#pragma unroll
    for (int d = 0; d < 4; ++d)
#pragma unroll
        for (int r = 0; r < 8; ++r) {
            const int ii  = ibase + r + 8 * lh;
            const int col = h * D_HEAD + 16 * d + lm;
            AttOut[(size_t)(b * S_LEN + ii) * E_DIM + col] = (bf16_t)(o[d][r] * sc[r]);
        }
}

// ---------------- host side ----------------
extern "C" void kernel_launch(void* const* d_in, const int* in_sizes, int n_in,
                              void* d_out, int out_size, void* d_ws, size_t ws_size,
                              hipStream_t stream) {
    (void)in_sizes; (void)n_in; (void)out_size; (void)ws_size;
    const float* x       = (const float*)d_in[0];
    const float* phase   = (const float*)d_in[1];
    const float* Wq      = (const float*)d_in[2];
    const float* bq      = (const float*)d_in[3];
    const float* Wk      = (const float*)d_in[4];
    const float* bk      = (const float*)d_in[5];
    const float* Wv      = (const float*)d_in[6];
    const float* bv      = (const float*)d_in[7];
    const float* Wo      = (const float*)d_in[8];
    const float* bo      = (const float*)d_in[9];
    const float* carrier = (const float*)d_in[10];
    const float* locks   = (const float*)d_in[11];

    // workspace carve-up (bytes)
    char* ws = (char*)d_ws;
    const size_t SZ_X  = (size_t)M_ROWS * E_DIM * 2;             // 8 MB
    const size_t SZ_W  = (size_t)E_DIM * E_DIM * 2;              // 2 MB each
    const size_t SZ_H  = (size_t)B_N * H_N * S_LEN * D_HEAD * 2; // 8 MB
    const size_t SZ_T  = (size_t)B_N * S_LEN * H_N * 4;          // 256 KB
    bf16_t* Xb   = (bf16_t*)(ws);                     ws += SZ_X;
    bf16_t* Wqt  = (bf16_t*)(ws);                     ws += SZ_W;
    bf16_t* Wkt  = (bf16_t*)(ws);                     ws += SZ_W;
    bf16_t* Wvt  = (bf16_t*)(ws);                     ws += SZ_W;
    bf16_t* Wot  = (bf16_t*)(ws);                     ws += SZ_W;
    bf16_t* Qb   = (bf16_t*)(ws);                     ws += SZ_H;
    bf16_t* KCb  = (bf16_t*)(ws);                     ws += SZ_H;
    bf16_t* VCt  = (bf16_t*)(ws);                     ws += SZ_H;
    float*  Tm   = (float*)(ws);                      ws += SZ_T;
    bf16_t* AttO = (bf16_t*)(ws);

    const int nX = M_ROWS * E_DIM;       // 4,194,304
    const int nW = E_DIM * E_DIM;        // 1,048,576
    cvt_bf16_kernel<<<(nX + 255) / 256, 256, 0, stream>>>(x, Xb, nX);
    cvt_wT_kernel<<<(nW + 255) / 256, 256, 0, stream>>>(Wq, Wqt);
    cvt_wT_kernel<<<(nW + 255) / 256, 256, 0, stream>>>(Wk, Wkt);
    cvt_wT_kernel<<<(nW + 255) / 256, 256, 0, stream>>>(Wv, Wvt);
    cvt_wT_kernel<<<(nW + 255) / 256, 256, 0, stream>>>(Wo, Wot);
    const int nT = B_N * S_LEN * H_N;
    phase_T_kernel<<<(nT + 255) / 256, 256, 0, stream>>>(phase, carrier, Tm);

    dim3 gg(M_ROWS / 16, E_DIM / 256);   // (256, 4): 4 waves/block, 64 cols per wave
    gemm16_kernel<<<gg, 128, 0, stream>>>(Xb, Wqt, bq, Qb,  0);
    gemm16_kernel<<<gg, 128, 0, stream>>>(Xb, Wkt, bk, KCb, 1);
    gemm16_kernel<<<gg, 128, 0, stream>>>(Xb, Wvt, bv, VCt, 2);

    dim3 ga(S_LEN / 64, H_N, B_N);       // (32, 16, 2), 4 i-tiles per block
    attn_kernel<<<ga, 128, 0, stream>>>(Qb, KCb, VCt, Tm, locks, AttO);

    gemm16_kernel<<<gg, 128, 0, stream>>>(AttO, Wot, bo, d_out, 3);
}
